// ImageToHypergraph_25477746000085
// MI455X (gfx1250) — matrix operand order repaired and checked by reading, not compiled, not run on hardware
//
#include <hip/hip_runtime.h>
#include <hip/hip_bf16.h>

typedef __attribute__((ext_vector_type(16))) _Float16 v16h;
typedef __attribute__((ext_vector_type(8)))  _Float16 v8h;
typedef __attribute__((ext_vector_type(8)))  float    v8f;

// ---------------- problem constants ----------------
#define BATCH   8
#define C1OUT   64
#define H1DIM   112
#define C2OUT   96        // D
#define HW2     56
#define NPTS    3136      // 56*56
#define NCLUS   10
#define KDIM2   576       // 9 taps * 64 channels (tap-major!)
#define NT_TILES 196      // 3136/16
#define MT_TILES 6        // 96/16
#define KS2     18        // 576/32
#define BN_EPS  1e-4f

// ---------------- workspace layout (bytes, 256-aligned) ----------------
#define OFFB_H1F16   ((size_t)0)                         // 8*112*112*64 f16 (NHWC)
#define OFFB_W2F16   ((size_t)12845056)                  // 96*576 f16 (m, tap, ic)
#define OFFB_PTSF32  ((size_t)12955648)                  // 8*3136*96 f32
#define OFFB_PTSF16  ((size_t)22589440)                  // 8*3136*96 f16 (N x D)
#define OFFB_PTST    ((size_t)27406336)                  // 8*96*3136 f16 (D x N)
#define OFFB_X2      ((size_t)32223232)                  // 8*3136 f32
#define OFFB_MB      ((size_t)32323584)                  // 8*3136*10 f32
#define OFFB_WT      ((size_t)33327104)                  // 8*16*3136 f16 (C x N), w = mb^2

// ---------------- output layout (floats) ----------------
#define OFF_HEM   0          // (B, C, N)  = 250880
#define OFF_PHI   250880     // (B, N, C)  = 250880
#define OFF_FEATS 501760     // (B, D, C)  = 7680
#define OFF_PP    509440     // (N, 2)     = 6272
#define OFF_HW    515712     // 2

__device__ __forceinline__ v8f wmma_f16(v16h a, v16h b, v8f c) {
  return __builtin_amdgcn_wmma_f32_16x16x32_f16(false, a, false, b, (short)0, c,
                                                false, false);
}
// A-fragment (16x32 f16): lane holds rows m, one contiguous 16-elem K run
__device__ __forceinline__ v16h loadA(const _Float16* __restrict__ row, int kb, int half) {
  v8h lo = *(const v8h*)(row + kb + half * 16);
  v8h hi = *(const v8h*)(row + kb + half * 16 + 8);
  return __builtin_shufflevector(lo, hi, 0, 1, 2, 3, 4, 5, 6, 7,
                                 8, 9, 10, 11, 12, 13, 14, 15);
}
// B-fragment (32x16 f16): lane holds one column, one contiguous 16-elem K run
__device__ __forceinline__ v16h loadB(const _Float16* __restrict__ col, int kb, int half) {
  return *(const v16h*)(col + kb + half * 16);
}

// ---------------- conv1 + bn1 + relu (direct f32 -> f16, NHWC out) ------------
__global__ void conv1_kernel(const float* __restrict__ x, const float* __restrict__ w,
                             const float* __restrict__ g, const float* __restrict__ be,
                             const float* __restrict__ mn, const float* __restrict__ vr,
                             _Float16* __restrict__ h1) {
  int i = blockIdx.x * 256 + threadIdx.x;           // NHWC linear index
  if (i >= BATCH * H1DIM * H1DIM * C1OUT) return;
  int oc = i & 63; int t = i >> 6;
  int ow = t % H1DIM; t /= H1DIM;
  int oh = t % H1DIM; int b = t / H1DIM;
  const float* xb = x + (size_t)b * 3 * 224 * 224;
  float acc = 0.f;
  for (int ic = 0; ic < 3; ++ic)
    for (int kh = 0; kh < 3; ++kh) {
      int ih = oh * 2 - 1 + kh; if (ih < 0 || ih >= 224) continue;
      for (int kw = 0; kw < 3; ++kw) {
        int iw = ow * 2 - 1 + kw; if (iw < 0 || iw >= 224) continue;
        acc += xb[(ic * 224 + ih) * 224 + iw] * w[((oc * 3 + ic) * 3 + kh) * 3 + kw];
      }
    }
  float inv = g[oc] * rsqrtf(vr[oc] + BN_EPS);
  float v = fmaxf(acc * inv + (be[oc] - mn[oc] * inv), 0.f);
  h1[i] = (_Float16)v;                              // NHWC
}

// ---------------- conv2 weight f32 -> f16, permute to (m, tap, ic) ------------
__global__ void prep_w2_kernel(const float* __restrict__ w, _Float16* __restrict__ o) {
  int i = blockIdx.x * 256 + threadIdx.x;
  if (i >= C2OUT * KDIM2) return;
  int ic = i & 63; int t = i >> 6;
  int tap = t % 9; int m = t / 9;
  int kh = tap / 3, kw = tap % 3;
  // conv2_w layout (O=96, I=64, KH=3, KW=3)
  o[i] = (_Float16)w[((m * 64 + ic) * 3 + kh) * 3 + kw];
}

// ---------------- conv2 implicit-GEMM WMMA + bn2 + relu ----------------
__global__ void conv2_wmma_kernel(const _Float16* __restrict__ h1,
                                  const _Float16* __restrict__ w2,
                                  const float* __restrict__ g, const float* __restrict__ be,
                                  const float* __restrict__ mn, const float* __restrict__ vr,
                                  float* __restrict__ ptsf32, _Float16* __restrict__ ptsf16,
                                  _Float16* __restrict__ ptsT) {
  int nt = blockIdx.x;           // spatial tile 0..195
  int mt = blockIdx.y;           // channel tile 0..5
  int b  = blockIdx.z;
  int lane = threadIdx.x & 31;
  int half = lane >> 4, l16 = lane & 15;
  const _Float16* h1b = h1 + (size_t)b * H1DIM * H1DIM * C1OUT;  // NHWC

  int n = nt * 16 + l16;         // B-fragment column (spatial)
  int oh = n / HW2, ow = n % HW2;
  int ohb = oh * 2 - 1, owb = ow * 2 - 1;
  const _Float16* wrow = w2 + (mt * 16 + l16) * KDIM2;  // A row (out channel)

  v8f acc = {};
  for (int ks = 0; ks < KS2; ++ks) {
    int kb = ks * 32;
    if (ks + 1 < KS2) __builtin_prefetch(wrow + kb + 32, 0, 3);
    v16h A = loadA(wrow, kb, half);
    // B: K ordered (tap, ic); run of 16 channels within one tap block
    int k0 = kb + half * 16;
    int tap = k0 >> 6;                 // 0..8
    int kh = tap / 3, kw = tap - (tap / 3) * 3;
    int ih = ohb + kh, iw = owb + kw;  // only low side can be OOB (pad=1, s=2)
    v16h B = {};
    if (((unsigned)ih < (unsigned)H1DIM) & ((unsigned)iw < (unsigned)H1DIM))
      B = *(const v16h*)(h1b + (((ih * H1DIM + iw) << 6) + (k0 & 63)));
    acc = wmma_f16(A, B, acc);
  }
#pragma unroll
  for (int r = 0; r < 8; ++r) {
    int oc = mt * 16 + r + 8 * half;
    int nn = nt * 16 + l16;
    float inv = g[oc] * rsqrtf(vr[oc] + BN_EPS);
    float val = fmaxf(acc[r] * inv + (be[oc] - mn[oc] * inv), 0.f);
    size_t pidx = ((size_t)b * NPTS + nn) * C2OUT + oc;
    ptsf32[pidx] = val;
    ptsf16[pidx] = (_Float16)val;
    ptsT[((size_t)b * C2OUT + oc) * NPTS + nn] = (_Float16)val;
  }
}

// ---------------- x2 = sum_d pts^2 ----------------
__global__ void x2_kernel(const float* __restrict__ pts, float* __restrict__ x2) {
  int i = blockIdx.x * 256 + threadIdx.x;
  if (i >= BATCH * NPTS) return;
  const float4* p = (const float4*)(pts + (size_t)i * C2OUT);
  float s = 0.f;
#pragma unroll
  for (int d = 0; d < C2OUT / 4; ++d) {
    float4 v = p[d];
    s += v.x * v.x + v.y * v.y + v.z * v.z + v.w * v.w;
  }
  x2[i] = s;
}

// ---------------- normalize initial memberships; seed wT = mb^2 (C x N) -----
__global__ void initmb_kernel(const float* __restrict__ m0, float* __restrict__ mb,
                              _Float16* __restrict__ wT) {
  int i = blockIdx.x * 256 + threadIdx.x;
  if (i >= BATCH * NPTS) return;
  int b = i / NPTS, n = i % NPTS;
  const float* p = m0 + (size_t)i * NCLUS;
  float s = 0.f;
  for (int c = 0; c < NCLUS; ++c) s += p[c];
  float r = 1.f / s;
  _Float16* wTb = wT + (size_t)b * 16 * NPTS;
  for (int c = 0; c < NCLUS; ++c) {
    float v = p[c] * r;
    mb[(size_t)i * NCLUS + c] = v;
    wTb[c * NPTS + n] = (_Float16)(v * v);
  }
  for (int c = NCLUS; c < 16; ++c) wTb[c * NPTS + n] = (_Float16)0.f;  // pad rows
}

// ---------------- persistent fuzzy-c-means (one block per batch) ----------------
__global__ __launch_bounds__(512)
void fcm_kernel(const _Float16* __restrict__ ptsf16, const _Float16* __restrict__ ptsT,
                const float* __restrict__ x2g, float* __restrict__ mbg,
                _Float16* __restrict__ wTg, float* __restrict__ out) {
  __shared__ float    c_raw[C2OUT * 16];   // raw WMMA accumulation (D x 16)
  __shared__ float    c_lds[C2OUT * 16];   // centers (D x 16), cols >=10 zero
  __shared__ _Float16 cf16T[16 * C2OUT];   // centers transposed (C x D), f16
  __shared__ float    sw[16];              // sum_n w[n][c]
  __shared__ float    cc[16];              // sum_d centers^2
  __shared__ float    red[1];              // diff norm^2 accumulator

  int b = blockIdx.x;
  int tid = threadIdx.x, lane = tid & 31, wave = tid >> 5;
  int half = lane >> 4, l16 = lane & 15;
  const _Float16* pB = ptsf16 + (size_t)b * NPTS * C2OUT;  // (N x D)
  const _Float16* pT = ptsT   + (size_t)b * C2OUT * NPTS;  // (D x N)
  const float*    x2b = x2g + b * NPTS;
  float*    mb  = mbg + (size_t)b * NPTS * NCLUS;
  _Float16* wTb = wTg + (size_t)b * 16 * NPTS;             // (16 x N) f16

  for (int it = 1;; ++it) {
    if (tid < 16) sw[tid] = 0.f;
    if (tid == 0) red[0] = 0.f;
    for (int i = tid; i < C2OUT * 16; i += 512) c_raw[i] = 0.f;
    __syncthreads();

    // ---- sw[c] = sum_n mb[n][c]^2  (m = 2 => w = mb^2)
    float loc[NCLUS];
#pragma unroll
    for (int c = 0; c < NCLUS; ++c) loc[c] = 0.f;
    for (int n = tid; n < NPTS; n += 512)
#pragma unroll
      for (int c = 0; c < NCLUS; ++c) { float v = mb[n * NCLUS + c]; loc[c] += v * v; }
#pragma unroll
    for (int c = 0; c < NCLUS; ++c) atomicAdd(&sw[c], loc[c]);
    __syncthreads();

    // ---- centers: (D x N) * (N x C) WMMA GEMM; 12 waves, K split in halves
    if (wave < 2 * MT_TILES) {
      int dtile = wave >> 1, khalf = wave & 1;
      const _Float16* arow = pT + (size_t)(dtile * 16 + l16) * NPTS;
      const _Float16* brow = wTb + (size_t)l16 * NPTS;
      v8f acc = {};
      for (int ks = khalf * 49; ks < khalf * 49 + 49; ++ks) {
        int kb = ks * 32;
        __builtin_prefetch(arow + kb + 256, 0, 3);
        v16h A = loadA(arow, kb, half);
        v16h B = loadB(brow, kb, half);   // w = mb^2, contiguous (C x N)
        acc = wmma_f16(A, B, acc);
      }
#pragma unroll
      for (int r = 0; r < 8; ++r)
        atomicAdd(&c_raw[(dtile * 16 + r + 8 * half) * 16 + l16], acc[r]);
    }
    __syncthreads();

    for (int i = tid; i < C2OUT * 16; i += 512) {
      int d = i >> 4, c = i & 15;
      float val = (c < NCLUS) ? c_raw[i] / sw[c] : 0.f;
      c_lds[i] = val;
      cf16T[c * C2OUT + d] = (_Float16)val;
    }
    __syncthreads();

    if (tid < 16) {
      float s = 0.f;
      for (int d = 0; d < C2OUT; ++d) { float v = c_lds[d * 16 + tid]; s += v * v; }
      cc[tid] = s;
    }
    __syncthreads();

    // ---- distances + membership update: (N x D) * (D x C) WMMA GEMM
    float dsum = 0.f;
    for (int nt = wave; nt < NT_TILES; nt += 16) {
      const _Float16* arow = pB + (size_t)(nt * 16 + l16) * C2OUT;
      const _Float16* brow = cf16T + l16 * C2OUT;
      v8f acc = {};
#pragma unroll
      for (int ks = 0; ks < 3; ++ks) {
        v16h A = loadA(arow, ks * 32, half);
        v16h B = loadB(brow, ks * 32, half);  // ds_load_b128 pairs
        acc = wmma_f16(A, B, acc);
      }
#pragma unroll
      for (int r = 0; r < 8; ++r) {
        int n = nt * 16 + r + 8 * half;
        int c = l16;
        float d2 = x2b[n] + cc[c] - 2.f * acc[r];
        // m=2: new = dist^-2 = 1/max(d2,1e-30)
        float u = (c < NCLUS) ? 1.f / fmaxf(d2, 1e-30f) : 0.f;
        float s = u;
        s += __shfl_xor(s, 1, 32);
        s += __shfl_xor(s, 2, 32);
        s += __shfl_xor(s, 4, 32);
        s += __shfl_xor(s, 8, 32);
        float nv = u / s;
        if (c < NCLUS) {
          float old = mb[n * NCLUS + c];
          float dd = nv - old;
          dsum += dd * dd;
          mb[n * NCLUS + c] = nv;
          wTb[c * NPTS + n] = (_Float16)(nv * nv);
        }
      }
    }
    atomicAdd(&red[0], dsum);
    __syncthreads();
    bool cont = (it < 1000) && (it < 2 || red[0] >= 1e-12f);
    __syncthreads();         // everyone has read red[0] before next-iter reset
    __threadfence_block();   // mb/wT stores visible block-wide before next read
    if (!cont) break;
  }

  // centers computed from the pre-final memberships == last c_lds contents
  for (int i = tid; i < C2OUT * NCLUS; i += 512) {
    int d = i / NCLUS, c = i % NCLUS;
    out[OFF_FEATS + ((size_t)b * C2OUT + d) * NCLUS + c] = c_lds[d * 16 + c];
  }
}

// ---------------- hyperedge_matrix (B, C, N): member point indices, -1 pad ----
__global__ void hem_kernel(const float* __restrict__ mbg, float* __restrict__ out) {
  int bc = blockIdx.x;                 // 0..79
  int b = bc / NCLUS, c = bc % NCLUS;
  float* o = out + OFF_HEM + (size_t)bc * NPTS;
  for (int i = threadIdx.x; i < NPTS; i += blockDim.x) o[i] = -1.f;
  __syncthreads();
  if (threadIdx.x == 0) {
    const float* mb = mbg + (size_t)b * NPTS * NCLUS;
    int s = 0;
    for (int n = 0; n < NPTS; ++n)
      if (mb[n * NCLUS + c] > 0.5f) o[s++] = (float)n;
  }
}

// ---------------- point_hyperedge_index (B, N, C) ----------------
__global__ void phi_kernel(const float* __restrict__ mbg, float* __restrict__ out) {
  int i = blockIdx.x * 256 + threadIdx.x;
  if (i >= BATCH * NPTS) return;
  const float* mb = mbg + (size_t)i * NCLUS;
  float* o = out + OFF_PHI + (size_t)i * NCLUS;
  int s = 0;
  for (int c = 0; c < NCLUS; ++c)
    if (mb[c] > 0.5f) o[s++] = (float)c;
  for (; s < NCLUS; ++s) o[s] = -1.f;
}

// ---------------- patch positions + (H, W) ----------------
__global__ void pos_kernel(float* __restrict__ out) {
  int n = blockIdx.x * 256 + threadIdx.x;
  if (n < NPTS) {
    out[OFF_PP + 2 * n + 0] = (float)(n / HW2);
    out[OFF_PP + 2 * n + 1] = (float)(n % HW2);
  }
  if (n == 0) { out[OFF_HW] = (float)HW2; out[OFF_HW + 1] = (float)HW2; }
}

extern "C" void kernel_launch(void* const* d_in, const int* in_sizes, int n_in,
                              void* d_out, int out_size, void* d_ws, size_t ws_size,
                              hipStream_t stream) {
  const float* x       = (const float*)d_in[0];
  const float* conv1_w = (const float*)d_in[1];
  const float* bn1_g   = (const float*)d_in[2];
  const float* bn1_b   = (const float*)d_in[3];
  const float* bn1_m   = (const float*)d_in[4];
  const float* bn1_v   = (const float*)d_in[5];
  const float* conv2_w = (const float*)d_in[6];
  const float* bn2_g   = (const float*)d_in[7];
  const float* bn2_b   = (const float*)d_in[8];
  const float* bn2_m   = (const float*)d_in[9];
  const float* bn2_v   = (const float*)d_in[10];
  const float* memb0   = (const float*)d_in[11];
  float* out = (float*)d_out;

  char* ws = (char*)d_ws;
  _Float16* h1f16  = (_Float16*)(ws + OFFB_H1F16);
  _Float16* w2f16  = (_Float16*)(ws + OFFB_W2F16);
  float*    ptsf32 = (float*)   (ws + OFFB_PTSF32);
  _Float16* ptsf16 = (_Float16*)(ws + OFFB_PTSF16);
  _Float16* ptsT   = (_Float16*)(ws + OFFB_PTST);
  float*    x2     = (float*)   (ws + OFFB_X2);
  float*    mbg    = (float*)   (ws + OFFB_MB);
  _Float16* wT     = (_Float16*)(ws + OFFB_WT);

  prep_w2_kernel<<<(C2OUT * KDIM2 + 255) / 256, 256, 0, stream>>>(conv2_w, w2f16);

  int n1 = BATCH * C1OUT * H1DIM * H1DIM;
  conv1_kernel<<<(n1 + 255) / 256, 256, 0, stream>>>(x, conv1_w, bn1_g, bn1_b, bn1_m,
                                                     bn1_v, h1f16);

  conv2_wmma_kernel<<<dim3(NT_TILES, MT_TILES, BATCH), 32, 0, stream>>>(
      h1f16, w2f16, bn2_g, bn2_b, bn2_m, bn2_v, ptsf32, ptsf16, ptsT);

  int np = BATCH * NPTS;
  x2_kernel<<<(np + 255) / 256, 256, 0, stream>>>(ptsf32, x2);
  initmb_kernel<<<(np + 255) / 256, 256, 0, stream>>>(memb0, mbg, wT);

  fcm_kernel<<<BATCH, 512, 0, stream>>>(ptsf16, ptsT, x2, mbg, wT, out);

  hem_kernel<<<BATCH * NCLUS, 256, 0, stream>>>(mbg, out);
  phi_kernel<<<(np + 255) / 256, 256, 0, stream>>>(mbg, out);
  pos_kernel<<<(NPTS + 255) / 256, 256, 0, stream>>>(out);
}
